// SimplifiedTransformer_45629732553210
// MI455X (gfx1250) — matrix operand order repaired and checked by reading, not compiled, not run on hardware
//
#include <hip/hip_runtime.h>
#include <stdint.h>

#define D_MODEL 1024
#define D_PRIME 256
#define N_LAYERS 4
#define BATCH 4
#define SEQ 2048
#define ROWS (BATCH * SEQ) /* 8192 */

typedef __attribute__((ext_vector_type(16))) __bf16 v16bf;
typedef __attribute__((ext_vector_type(8))) float v8f;

union ABPack { uint32_t u[8]; v16bf v; };

__device__ __forceinline__ uint16_t f2bf(float f) {
  union { float f; uint32_t u; } c; c.f = f;
  uint32_t u = c.u;
  u += 0x7FFFu + ((u >> 16) & 1u);      // round-to-nearest-even
  return (uint16_t)(u >> 16);
}

// A operand: 16x32 bf16 tile, row-major source, row = m0 + (lane&15).
// ISA layout: lane half hi holds K = {hi*8..hi*8+7, 16+hi*8..16+hi*8+7} as dword pairs.
__device__ __forceinline__ v16bf load_a(const uint32_t* row_dw, int hi) {
  ABPack p;
#pragma unroll
  for (int r = 0; r < 4; ++r) {
    p.u[r]     = row_dw[hi * 4 + r];
    p.u[4 + r] = row_dw[8 + hi * 4 + r];
  }
  return p.v;
}

// B operand: 32x16 bf16 tile from an N-major (n-row, K-contiguous) source,
// n = n0 + (lane&15). Lane half hi holds K = {hi*8..hi*8+7, 16+hi*8..16+hi*8+7} as dword pairs.
__device__ __forceinline__ v16bf load_b(const uint32_t* row_dw, int hi) {
  ABPack p;
#pragma unroll
  for (int r = 0; r < 8; ++r) p.u[r] = row_dw[(r >> 2) * 8 + hi * 4 + (r & 3)];
  return p.v;
}

__device__ __forceinline__ v8f wmma_bf16(v16bf a, v16bf b, v8f c) {
  return __builtin_amdgcn_wmma_f32_16x16x32_bf16(false, a, false, b, (short)0, c,
                                                 false, false);
}

// ---------------------------------------------------------------- weights->bf16 (transposed)
__global__ __launch_bounds__(256) void k_convert_weights(
    const float* __restrict__ Wq, const float* __restrict__ Wk,
    const float* __restrict__ Wv, const float* __restrict__ Wo,
    uint16_t* __restrict__ WqT, uint16_t* __restrict__ WkT,
    uint16_t* __restrict__ WvT, uint16_t* __restrict__ WoT) {
  uint32_t t = blockIdx.x * blockDim.x + threadIdx.x;   // 4 * 2^20 threads
  uint32_t tensor = t >> 20;
  uint32_t r = t & 0xFFFFFu;
  uint32_t l = r >> 18;               // layer
  uint32_t rem = r & 0x3FFFFu;        // 2^18 = 1024*256
  if (tensor < 3) {
    // Wq/Wk/Wv: [L][1024][256] -> T: [L][256][1024]  (n-major, K=d contiguous)
    uint32_t d = rem >> 8, p = rem & 255u;
    const float* src = (tensor == 0) ? Wq : (tensor == 1) ? Wk : Wv;
    uint16_t* dst = (tensor == 0) ? WqT : (tensor == 1) ? WkT : WvT;
    dst[l * 262144u + p * 1024u + d] = f2bf(src[l * 262144u + d * 256u + p]);
  } else {
    // Wo: [L][256][1024] -> T: [L][1024][256]  (n=d major, K=p contiguous)
    uint32_t p = rem >> 10, d = rem & 1023u;
    WoT[l * 262144u + d * 256u + p] = f2bf(Wo[l * 262144u + p * 1024u + d]);
  }
}

// ---------------------------------------------------------------- embedding gather
__global__ __launch_bounds__(256) void k_embed(
    const int* __restrict__ x, const float* __restrict__ W_enc,
    float* __restrict__ Xf, uint16_t* __restrict__ Xbf) {
  uint32_t t = blockIdx.x * blockDim.x + threadIdx.x;   // ROWS * 256 (float4 granules)
  uint32_t row = t >> 8, q = t & 255u;
  int tok = x[row];
  float4 w = ((const float4*)W_enc)[(size_t)tok * 256u + q];
  ((float4*)Xf)[(size_t)row * 256u + q] = w;
  uint2 pk;
  pk.x = (uint32_t)f2bf(w.x) | ((uint32_t)f2bf(w.y) << 16);
  pk.y = (uint32_t)f2bf(w.z) | ((uint32_t)f2bf(w.w) << 16);
  ((uint2*)Xbf)[(size_t)row * 256u + q] = pk;
}

// ---------------------------------------------------------------- Q/K/V projections (2x2 tiles)
__global__ __launch_bounds__(256) void k_proj(
    const uint16_t* __restrict__ Xbf,
    const uint16_t* __restrict__ WqT, const uint16_t* __restrict__ WkT,
    const uint16_t* __restrict__ WvT,
    uint16_t* __restrict__ Qbf, uint16_t* __restrict__ Kbf,
    uint16_t* __restrict__ Vbf, int layer) {
  int t = blockIdx.x * (blockDim.x >> 5) + (threadIdx.x >> 5);
  if (t >= 3 * 256 * 8) return;            // 3 weights * 256 m-pairs * 8 n-pairs
  int w = t / 2048;
  int r = t - w * 2048;
  int m0 = (r >> 3) * 32;                  // 32-row super-tile
  int n0 = (r & 7) * 32;                   // 32-col super-tile
  int lane = threadIdx.x & 31, col = lane & 15, hi = lane >> 4;

  const uint16_t* WT = ((w == 0) ? WqT : (w == 1) ? WkT : WvT) + (size_t)layer * 262144u;
  uint16_t* Out = (w == 0) ? Qbf : (w == 1) ? Kbf : Vbf;

  const uint32_t* Arow0 = (const uint32_t*)(Xbf + (size_t)(m0 + col) * D_MODEL);
  const uint32_t* Arow1 = (const uint32_t*)(Xbf + (size_t)(m0 + 16 + col) * D_MODEL);
  const uint32_t* Brow0 = (const uint32_t*)(WT + (size_t)(n0 + col) * D_MODEL);
  const uint32_t* Brow1 = (const uint32_t*)(WT + (size_t)(n0 + 16 + col) * D_MODEL);

  v8f acc00 = {}, acc01 = {}, acc10 = {}, acc11 = {};
#pragma unroll 2
  for (int kk = 0; kk < D_MODEL; kk += 32) {
    v16bf A0 = load_a(Arow0 + (kk >> 1), hi);
    v16bf A1 = load_a(Arow1 + (kk >> 1), hi);
    v16bf B0 = load_b(Brow0 + (kk >> 1), hi);
    v16bf B1 = load_b(Brow1 + (kk >> 1), hi);
    acc00 = wmma_bf16(A0, B0, acc00);      // 4 independent accumulation chains
    acc01 = wmma_bf16(A0, B1, acc01);
    acc10 = wmma_bf16(A1, B0, acc10);
    acc11 = wmma_bf16(A1, B1, acc11);
  }
#pragma unroll
  for (int v = 0; v < 8; ++v) {
    int mA = m0 + v + hi * 8, mB = mA + 16;
    int nA = n0 + col, nB = nA + 16;
    Out[(size_t)mA * D_PRIME + nA] = f2bf(acc00[v]);
    Out[(size_t)mA * D_PRIME + nB] = f2bf(acc01[v]);
    Out[(size_t)mB * D_PRIME + nA] = f2bf(acc10[v]);
    Out[(size_t)mB * D_PRIME + nB] = f2bf(acc11[v]);
  }
}

// ------------------------------------------ fused Q.K^T + argmax (Q tile held in registers)
__global__ __launch_bounds__(256) void k_score_argmax(
    const uint16_t* __restrict__ Qbf, const uint16_t* __restrict__ Kbf,
    int* __restrict__ idx) {
  int t = blockIdx.x * (blockDim.x >> 5) + (threadIdx.x >> 5);
  if (t >= BATCH * (SEQ / 16)) return;           // 512 waves
  int b = t >> 7, qt = t & 127;
  int lane = threadIdx.x & 31, col = lane & 15, hi = lane >> 4;

  // Hoist the whole 16x256 Q tile (8 A operands, 64 VGPRs/lane): steady-state loop
  // then issues only B loads (2 x B128 per WMMA).
  const uint32_t* Arow = (const uint32_t*)(Qbf + (size_t)(b * SEQ + qt * 16 + col) * D_PRIME);
  v16bf Areg[8];
#pragma unroll
  for (int j = 0; j < 8; ++j) Areg[j] = load_a(Arow + j * 16, hi);

  float bestS[8];
  int bestK[8];
#pragma unroll
  for (int v = 0; v < 8; ++v) { bestS[v] = -3.0e38f; bestK[v] = 0; }

  for (int kt = 0; kt < SEQ / 16; kt += 2) {     // two key tiles per iteration -> WMMA ILP
    const uint32_t* Brow0 =
        (const uint32_t*)(Kbf + (size_t)(b * SEQ + kt * 16 + col) * D_PRIME);
    const uint32_t* Brow1 =
        (const uint32_t*)(Kbf + (size_t)(b * SEQ + (kt + 1) * 16 + col) * D_PRIME);
    v8f acc0 = {}, acc1 = {};
#pragma unroll
    for (int j = 0; j < 8; ++j) {
      v16bf B0 = load_b(Brow0 + j * 16, hi);
      v16bf B1 = load_b(Brow1 + j * 16, hi);
      acc0 = wmma_bf16(Areg[j], B0, acc0);
      acc1 = wmma_bf16(Areg[j], B1, acc1);
    }
    int key0 = kt * 16 + col;                    // keys in acc1 tile are all > acc0 tile
#pragma unroll
    for (int v = 0; v < 8; ++v) {
      float s0 = acc0[v];
      if (s0 > bestS[v]) { bestS[v] = s0; bestK[v] = key0; }     // strict > keeps earliest
      float s1 = acc1[v];
      if (s1 > bestS[v]) { bestS[v] = s1; bestK[v] = key0 + 16; }
    }
  }
  // Row-wise reduction across the 16 lanes of each half (rows live in lane halves).
#pragma unroll
  for (int v = 0; v < 8; ++v) {
#pragma unroll
    for (int m = 1; m <= 8; m <<= 1) {
      float oS = __shfl_xor(bestS[v], m, 32);
      int oK = __shfl_xor(bestK[v], m, 32);
      if (oS > bestS[v] || (oS == bestS[v] && oK < bestK[v])) {
        bestS[v] = oS; bestK[v] = oK;
      }
    }
  }
  if (col == 0) {
#pragma unroll
    for (int v = 0; v < 8; ++v) idx[b * SEQ + qt * 16 + v + hi * 8] = bestK[v];
  }
}

// ---------------------------------- gather(V) @ Wo + relu + residual (2x2 tiles)
__global__ __launch_bounds__(256) void k_out(
    const uint16_t* __restrict__ Vbf, const uint16_t* __restrict__ WoT,
    const int* __restrict__ idx, float* __restrict__ Xf, uint16_t* __restrict__ Xbf,
    int layer) {
  int t = blockIdx.x * (blockDim.x >> 5) + (threadIdx.x >> 5);
  if (t >= 256 * 32) return;                 // 256 m-pairs * 32 n-pairs
  int m0 = (t >> 5) * 32;
  int n0 = (t & 31) * 32;
  int b = m0 >> 11;                          // batch (32-row tile never crosses a batch)
  int lane = threadIdx.x & 31, col = lane & 15, hi = lane >> 4;

  int ridx0 = idx[m0 + col];                 // argmax attention gather (per batch)
  int ridx1 = idx[m0 + 16 + col];
  const uint32_t* Arow0 = (const uint32_t*)(Vbf + (size_t)(b * SEQ + ridx0) * D_PRIME);
  const uint32_t* Arow1 = (const uint32_t*)(Vbf + (size_t)(b * SEQ + ridx1) * D_PRIME);
  const uint16_t* WoL = WoT + (size_t)layer * 262144u;
  const uint32_t* Brow0 = (const uint32_t*)(WoL + (size_t)(n0 + col) * D_PRIME);
  const uint32_t* Brow1 = (const uint32_t*)(WoL + (size_t)(n0 + 16 + col) * D_PRIME);

  v8f acc00 = {}, acc01 = {}, acc10 = {}, acc11 = {};
#pragma unroll
  for (int kk = 0; kk < D_PRIME; kk += 32) {
    v16bf A0 = load_a(Arow0 + (kk >> 1), hi);
    v16bf A1 = load_a(Arow1 + (kk >> 1), hi);
    v16bf B0 = load_b(Brow0 + (kk >> 1), hi);
    v16bf B1 = load_b(Brow1 + (kk >> 1), hi);
    acc00 = wmma_bf16(A0, B0, acc00);
    acc01 = wmma_bf16(A0, B1, acc01);
    acc10 = wmma_bf16(A1, B0, acc10);
    acc11 = wmma_bf16(A1, B1, acc11);
  }
#pragma unroll
  for (int v = 0; v < 8; ++v) {
    int mA = m0 + v + hi * 8, mB = mA + 16;
    int nA = n0 + col, nB = nA + 16;
    size_t oAA = (size_t)mA * D_MODEL + nA, oAB = (size_t)mA * D_MODEL + nB;
    size_t oBA = (size_t)mB * D_MODEL + nA, oBB = (size_t)mB * D_MODEL + nB;
    float vAA = fmaxf(acc00[v], 0.0f) + Xf[oAA];   // relu(out) + residual
    float vAB = fmaxf(acc01[v], 0.0f) + Xf[oAB];
    float vBA = fmaxf(acc10[v], 0.0f) + Xf[oBA];
    float vBB = fmaxf(acc11[v], 0.0f) + Xf[oBB];
    Xf[oAA] = vAA; Xbf[oAA] = f2bf(vAA);
    Xf[oAB] = vAB; Xbf[oAB] = f2bf(vAB);
    Xf[oBA] = vBA; Xbf[oBA] = f2bf(vBA);
    Xf[oBB] = vBB; Xbf[oBB] = f2bf(vBB);
  }
}

// ---------------------------------------------------------------- final readout
__global__ void k_final(const float* __restrict__ Xf, float* __restrict__ out,
                        int out_size) {
  int b = threadIdx.x;
  if (b < BATCH) {
    float raw = Xf[(size_t)(b * SEQ + (SEQ - 1)) * D_MODEL + 0];
    if (b < out_size) out[b] = raw;
    if (BATCH + b < out_size) out[BATCH + b] = (raw > 0.0f) ? 1.0f : 0.0f;
  }
}

extern "C" void kernel_launch(void* const* d_in, const int* in_sizes, int n_in,
                              void* d_out, int out_size, void* d_ws, size_t ws_size,
                              hipStream_t stream) {
  const int* x = (const int*)d_in[0];
  const float* W_enc = (const float*)d_in[1];
  const float* Wq = (const float*)d_in[2];
  const float* Wk = (const float*)d_in[3];
  const float* Wv = (const float*)d_in[4];
  const float* Wo = (const float*)d_in[5];

  char* ws = (char*)d_ws;
  float* Xf = (float*)ws;        ws += (size_t)ROWS * D_MODEL * 4;   // 32 MB
  uint16_t* Xbf = (uint16_t*)ws; ws += (size_t)ROWS * D_MODEL * 2;   // 16 MB
  uint16_t* Qbf = (uint16_t*)ws; ws += (size_t)ROWS * D_PRIME * 2;   // 4 MB
  uint16_t* Kbf = (uint16_t*)ws; ws += (size_t)ROWS * D_PRIME * 2;   // 4 MB
  uint16_t* Vbf = (uint16_t*)ws; ws += (size_t)ROWS * D_PRIME * 2;   // 4 MB
  int* idx = (int*)ws;           ws += (size_t)ROWS * 4;             // 32 KB
  uint16_t* WqT = (uint16_t*)ws; ws += (size_t)N_LAYERS * 262144 * 2;
  uint16_t* WkT = (uint16_t*)ws; ws += (size_t)N_LAYERS * 262144 * 2;
  uint16_t* WvT = (uint16_t*)ws; ws += (size_t)N_LAYERS * 262144 * 2;
  uint16_t* WoT = (uint16_t*)ws; ws += (size_t)N_LAYERS * 262144 * 2;

  k_convert_weights<<<16384, 256, 0, stream>>>(Wq, Wk, Wv, Wo, WqT, WkT, WvT, WoT);
  k_embed<<<ROWS, 256, 0, stream>>>(x, W_enc, Xf, Xbf);

  for (int l = 0; l < N_LAYERS; ++l) {
    k_proj<<<768, 256, 0, stream>>>(Xbf, WqT, WkT, WvT, Qbf, Kbf, Vbf, l);   // 6144 waves
    k_score_argmax<<<64, 256, 0, stream>>>(Qbf, Kbf, idx);                    // 512 waves
    k_out<<<1024, 256, 0, stream>>>(Vbf, WoT, idx, Xf, Xbf, l);               // 8192 waves
  }

  k_final<<<1, 32, 0, stream>>>(Xf, (float*)d_out, out_size);
}
